// NRMSModel_4277787427703
// MI455X (gfx1250) — matrix operand order repaired and checked by reading, not compiled, not run on hardware
//
#include <hip/hip_runtime.h>
#include <hip/hip_bf16.h>
#include <stdint.h>

// ---------------- problem constants ----------------
#define DIM   256     // D (head/emb dim)
#define TLEN  64      // title length T
#define HISTN 50      // history length
#define BATCH 1024    // batch B

// Flip to 0 if the async-to-LDS mnemonic is rejected by the assembler.
#ifndef USE_ASYNC_LDS
#define USE_ASYNC_LDS 1
#endif

typedef __bf16 bf16_t;
typedef __attribute__((ext_vector_type(16))) __bf16 v16bf;
typedef __attribute__((ext_vector_type(8)))  float  v8f;

union AFrag { v16bf v; uint32_t u[8]; bf16_t h[16]; };
union BFrag { v16bf v; uint4 q4[2]; uint32_t u[8]; bf16_t h[16]; };

__device__ __forceinline__ int imin(int a, int b) { return a < b ? a : b; }

// 16-bit A-matrix 16x32 layout (contiguous, matches B frags): VGPR j holds K = 16*half + 2*j (+0,+1)
__device__ __forceinline__ int a_koff(int j, int half) {
  return (half << 4) + (j << 1);
}

// ---- CDNA5 async global->LDS copy (ASYNCcnt-tracked, ISA 15.18.3 op 98) ----
__device__ __forceinline__ void async_copy_b128(const void* g, void* lds) {
#if USE_ASYNC_LDS
  const uint64_t ga = (uint64_t)(uintptr_t)g;
  // generic pointer to __shared__: low 32 bits are the LDS byte offset
  const uint32_t la = (uint32_t)(uintptr_t)lds;
  asm volatile("global_load_async_to_lds_b128 %0, %1, off"
               :: "v"(la), "v"(ga) : "memory");
#else
  *(uint4*)lds = *(const uint4*)g;
#endif
}
__device__ __forceinline__ void async_wait0() {
#if USE_ASYNC_LDS
  asm volatile("s_wait_asynccnt 0" ::: "memory");
#endif
}

// =====================================================================
// GEMM: Out[M,N] = A[M,K](bf16) * Bt[N,K](bf16)^T + bias
// block = 256 threads = 8 waves; each wave -> full 16(M) x 256(N) strip
// (16 v8f accumulators) so the big A stream is read from HBM exactly once.
// B chunk [256 x 32] is staged into LDS via async-to-LDS, read as ds_load_b128.
// grid = (M/128, N/256). K mult of 32, N mult of 256, M mult of 128.
// =====================================================================
template<bool OUT_F32>
__global__ void __launch_bounds__(256)
gemm_bf16(const bf16_t* __restrict__ A,
          const bf16_t* __restrict__ Bt,
          const float*  __restrict__ bias,
          void*         __restrict__ Out,
          int M, int N, int K)
{
  __shared__ bf16_t ldsB[256 * 32];   // 16 KB: [n within chunk][k within chunk]

  const int tid  = threadIdx.x;
  const int lane = tid & 31;
  const int wave = tid >> 5;
  const int half = lane >> 4;
  const int l16  = lane & 15;
  const int row0 = blockIdx.x * 128 + wave * 16;
  const int col0 = blockIdx.y * 256;

  v8f acc[16];
  #pragma unroll
  for (int t = 0; t < 16; ++t)
    #pragma unroll
    for (int i = 0; i < 8; ++i) acc[t][i] = 0.0f;

  const bf16_t* arow = A + (size_t)(row0 + l16) * K;
  const int srow = tid >> 2;   // staging: 4 threads per 64-byte row
  const int sseg = tid & 3;    // 16-byte segment

  for (int k0 = 0; k0 < K; k0 += 32) {
    // ---- cooperative async stage of Bt[col0..col0+255][k0..k0+31] ----
    #pragma unroll
    for (int rr = 0; rr < 4; ++rr) {
      const int n = rr * 64 + srow;
      async_copy_b128(Bt + (size_t)(col0 + n) * K + k0 + sseg * 8,
                      &ldsB[n * 32 + sseg * 8]);
    }
    if (k0 + 32 < K) __builtin_prefetch(arow + k0 + 32, 0, 1);  // global_prefetch_b8

    AFrag a;
    #pragma unroll
    for (int j = 0; j < 8; ++j)
      a.u[j] = *(const uint32_t*)(arow + k0 + a_koff(j, half));

    async_wait0();        // my async writes done
    __syncthreads();      // everyone's async writes done

    #pragma unroll
    for (int t = 0; t < 16; ++t) {
      const bf16_t* lb = &ldsB[(t * 16 + l16) * 32 + half * 16];
      BFrag b;
      b.q4[0] = *(const uint4*)(lb);
      b.q4[1] = *(const uint4*)(lb + 8);
      acc[t] = __builtin_amdgcn_wmma_f32_16x16x32_bf16(
          false, a.v, false, b.v, (short)0, acc[t], false, false);
    }
    __syncthreads();      // protect LDS before next stage
  }

  #pragma unroll
  for (int t = 0; t < 16; ++t) {
    const int n = col0 + t * 16 + l16;
    const float bv = bias ? bias[n] : 0.0f;
    #pragma unroll
    for (int r = 0; r < 8; ++r) {
      const int m = row0 + r + half * 8;
      const float val = acc[t][r] + bv;
      if (OUT_F32) ((float*)Out)[(size_t)m * N + n] = val;
      else         ((bf16_t*)Out)[(size_t)m * N + n] = (bf16_t)val;
    }
  }
}

// =====================================================================
// Fused self-attention: per (batch, 16-row tile). S padded to 64.
// scores = QK^T/16 -> masked softmax -> P@V. All matmuls via WMMA.
// block = 32 threads (1 wave). grid = (4, B).
// =====================================================================
template<int S_VALID>
__global__ void __launch_bounds__(32)
attention_kernel(const bf16_t* __restrict__ Q,
                 const bf16_t* __restrict__ Km,
                 const bf16_t* __restrict__ V,
                 float* __restrict__ Out,
                 float scale)
{
  constexpr int D  = DIM;
  constexpr int SP = 64;
  __shared__ bf16_t p_lds[16 * SP];   // probability tile, A-frag source

  const int lane = threadIdx.x & 31;
  const int half = lane >> 4;
  const int l16  = lane & 15;
  const int b    = blockIdx.y;
  const int rt   = blockIdx.x;              // row tile 0..3
  const int qrow = imin(rt * 16 + l16, S_VALID - 1);
  const bf16_t* qbase = Q + ((size_t)b * S_VALID + qrow) * D;

  // ---- scores = Q K^T ----
  v8f sacc[4];
  #pragma unroll
  for (int t = 0; t < 4; ++t)
    #pragma unroll
    for (int i = 0; i < 8; ++i) sacc[t][i] = 0.0f;

  for (int k0 = 0; k0 < D; k0 += 32) {
    AFrag a;
    #pragma unroll
    for (int j = 0; j < 8; ++j)
      a.u[j] = *(const uint32_t*)(qbase + k0 + a_koff(j, half));
    #pragma unroll
    for (int t = 0; t < 4; ++t) {
      const int trow = imin(t * 16 + l16, S_VALID - 1);
      const bf16_t* kb = Km + ((size_t)b * S_VALID + trow) * D + k0 + half * 16;
      BFrag bb;
      bb.q4[0] = *(const uint4*)(kb);
      bb.q4[1] = *(const uint4*)(kb + 8);
      sacc[t] = __builtin_amdgcn_wmma_f32_16x16x32_bf16(
          false, a.v, false, bb.v, (short)0, sacc[t], false, false);
    }
  }

  // ---- scale + column mask ----
  #pragma unroll
  for (int t = 0; t < 4; ++t) {
    const int col = t * 16 + l16;
    #pragma unroll
    for (int r = 0; r < 8; ++r) {
      float s = sacc[t][r] * scale;
      if (col >= S_VALID) s = -1e30f;
      sacc[t][r] = s;
    }
  }

  // ---- softmax over the 64 key columns (16 lanes x 4 tiles per row) ----
  #pragma unroll
  for (int r = 0; r < 8; ++r) {
    float mx = sacc[0][r];
    #pragma unroll
    for (int t = 1; t < 4; ++t) mx = fmaxf(mx, sacc[t][r]);
    #pragma unroll
    for (int o = 1; o <= 8; o <<= 1) mx = fmaxf(mx, __shfl_xor(mx, o, 32));
    float sm = 0.0f;
    #pragma unroll
    for (int t = 0; t < 4; ++t) {
      const float e = __expf(sacc[t][r] - mx);
      sacc[t][r] = e;
      sm += e;
    }
    #pragma unroll
    for (int o = 1; o <= 8; o <<= 1) sm += __shfl_xor(sm, o, 32);
    const float inv = 1.0f / sm;
    const int m = r + half * 8;          // D-layout row
    #pragma unroll
    for (int t = 0; t < 4; ++t)
      p_lds[m * SP + t * 16 + l16] = (bf16_t)(sacc[t][r] * inv);
  }
  __syncthreads();

  // ---- out = P @ V ----
  for (int nc = 0; nc < 4; ++nc) {
    const int n0 = nc * 64;
    v8f oacc[4];
    #pragma unroll
    for (int t = 0; t < 4; ++t)
      #pragma unroll
      for (int i = 0; i < 8; ++i) oacc[t][i] = 0.0f;

    for (int k0 = 0; k0 < SP; k0 += 32) {
      AFrag a;
      #pragma unroll
      for (int j = 0; j < 8; ++j)
        a.u[j] = *(const uint32_t*)(&p_lds[l16 * SP + k0 + a_koff(j, half)]);
      #pragma unroll
      for (int t = 0; t < 4; ++t) {
        const int n = n0 + t * 16 + l16;
        BFrag bb;
        #pragma unroll
        for (int e = 0; e < 16; ++e) {
          const int srow = imin(k0 + half * 16 + e, S_VALID - 1);
          bb.h[e] = V[((size_t)b * S_VALID + srow) * D + n];
        }
        oacc[t] = __builtin_amdgcn_wmma_f32_16x16x32_bf16(
            false, a.v, false, bb.v, (short)0, oacc[t], false, false);
      }
    }
    #pragma unroll
    for (int t = 0; t < 4; ++t) {
      #pragma unroll
      for (int r = 0; r < 8; ++r) {
        const int m = rt * 16 + r + half * 8;
        if (m < S_VALID)
          Out[((size_t)b * S_VALID + m) * D + n0 + t * 16 + l16] = oacc[t][r];
      }
    }
  }
}

// =====================================================================
// Elementwise / data-movement helpers
// =====================================================================
__global__ void f32_to_bf16(const float* __restrict__ in, bf16_t* __restrict__ out, size_t n) {
  const size_t i = (size_t)blockIdx.x * blockDim.x + threadIdx.x;
  if (i < n) out[i] = (bf16_t)in[i];
}

// in[K,N] f32  ->  out[N,K] bf16 (transposed weights for contiguous B-frag loads)
__global__ void transpose_to_bf16(const float* __restrict__ in, bf16_t* __restrict__ out,
                                  int K, int N) {
  const int i = blockIdx.x * blockDim.x + threadIdx.x;
  if (i >= K * N) return;
  const int k = i / N, n = i % N;
  out[(size_t)n * K + k] = (bf16_t)in[i];
}

// embedding gather + f32->bf16; one row (256 elems) per block of 64 threads
__global__ void gather_embed(const int* __restrict__ idx, const float* __restrict__ table,
                             bf16_t* __restrict__ out, int nrows) {
  const int row = blockIdx.x;
  if (row >= nrows) return;
  const int t = threadIdx.x;
  const float4 v = *(const float4*)(table + (size_t)idx[row] * DIM + t * 4);
  bf16_t* o = out + (size_t)row * DIM + t * 4;
  o[0] = (bf16_t)v.x; o[1] = (bf16_t)v.y; o[2] = (bf16_t)v.z; o[3] = (bf16_t)v.w;
}

// attention pooling: rep[b,:] = sum_s softmax_s(x[b,s,:].aw + ab) * x[b,s,:]
template<int S>
__global__ void __launch_bounds__(256)
attn_pool(const float* __restrict__ X, const float* __restrict__ aw,
          const float* __restrict__ ab, float* __restrict__ rep,
          bf16_t* __restrict__ rep_bf) {
  __shared__ float sc[S];
  __shared__ float red[2];
  const int b = blockIdx.x;
  const int t = threadIdx.x;
  const float* xb = X + (size_t)b * S * DIM;
  if (t < S) {
    float s = ab[0];
    for (int d = 0; d < DIM; ++d) s += xb[t * DIM + d] * aw[d];
    sc[t] = s;
  }
  __syncthreads();
  if (t == 0) {
    float m = sc[0];
    for (int i = 1; i < S; ++i) m = fmaxf(m, sc[i]);
    float s = 0.0f;
    for (int i = 0; i < S; ++i) s += __expf(sc[i] - m);
    red[0] = m; red[1] = s;
  }
  __syncthreads();
  const float m = red[0], inv = 1.0f / red[1];
  float acc = 0.0f;
  for (int i = 0; i < S; ++i) acc += xb[i * DIM + t] * (__expf(sc[i] - m) * inv);
  rep[(size_t)b * DIM + t] = acc;
  rep_bf[(size_t)b * DIM + t] = (bf16_t)acc;
}

// row-wise softmax over N columns; one block (256 threads) per row
__global__ void __launch_bounds__(256)
row_softmax(const float* __restrict__ in, float* __restrict__ out, int N) {
  __shared__ float red[256];
  const int row = blockIdx.x;
  const int t = threadIdx.x;
  const float* r = in + (size_t)row * N;
  float m = -1e30f;
  for (int i = t; i < N; i += 256) m = fmaxf(m, r[i]);
  red[t] = m; __syncthreads();
  for (int o = 128; o > 0; o >>= 1) { if (t < o) red[t] = fmaxf(red[t], red[t + o]); __syncthreads(); }
  m = red[0]; __syncthreads();
  float s = 0.0f;
  for (int i = t; i < N; i += 256) s += __expf(r[i] - m);
  red[t] = s; __syncthreads();
  for (int o = 128; o > 0; o >>= 1) { if (t < o) red[t] += red[t + o]; __syncthreads(); }
  const float inv = 1.0f / red[0];
  for (int i = t; i < N; i += 256) out[(size_t)row * N + i] = __expf(r[i] - m) * inv;
}

// =====================================================================
// Host launcher
// =====================================================================
extern "C" void kernel_launch(void* const* d_in, const int* in_sizes, int n_in,
                              void* d_out, int out_size, void* d_ws, size_t ws_size,
                              hipStream_t stream) {
  (void)in_sizes; (void)n_in; (void)out_size; (void)ws_size;

  const float* clicked = (const float*)d_in[0];
  const int*   cand    = (const int*)  d_in[1];
  const float* emb     = (const float*)d_in[2];
  const float* Wq_n = (const float*)d_in[3];  const float* bq_n = (const float*)d_in[4];
  const float* Wk_n = (const float*)d_in[5];  const float* bk_n = (const float*)d_in[6];
  const float* Wv_n = (const float*)d_in[7];  const float* bv_n = (const float*)d_in[8];
  const float* aw_n = (const float*)d_in[9];  const float* ab_n = (const float*)d_in[10];
  const float* W_u  = (const float*)d_in[11]; const float* b_u  = (const float*)d_in[12];
  const float* Wq_u = (const float*)d_in[13]; const float* bq_u = (const float*)d_in[14];
  const float* Wk_u = (const float*)d_in[15]; const float* bk_u = (const float*)d_in[16];
  const float* Wv_u = (const float*)d_in[17]; const float* bv_u = (const float*)d_in[18];
  const float* aw_u = (const float*)d_in[19]; const float* ab_u = (const float*)d_in[20];

  const size_t MU = (size_t)BATCH * HISTN;   // 51200 user rows
  const size_t MN = (size_t)BATCH * TLEN;    // 65536 news rows

  char* ws = (char*)d_ws;
  size_t off = 0;
  auto alloc = [&](size_t bytes) -> char* {
    char* p = ws + off;
    off += (bytes + 255) & ~(size_t)255;
    return p;
  };

  bf16_t* wqn_t = (bf16_t*)alloc((size_t)DIM * DIM * 2);
  bf16_t* wkn_t = (bf16_t*)alloc((size_t)DIM * DIM * 2);
  bf16_t* wvn_t = (bf16_t*)alloc((size_t)DIM * DIM * 2);
  bf16_t* wqu_t = (bf16_t*)alloc((size_t)DIM * DIM * 2);
  bf16_t* wku_t = (bf16_t*)alloc((size_t)DIM * DIM * 2);
  bf16_t* wvu_t = (bf16_t*)alloc((size_t)DIM * DIM * 2);
  bf16_t* wu_t  = (bf16_t*)alloc((size_t)DIM * TLEN * 2);
  bf16_t* clicked_bf = (bf16_t*)alloc(3 * MU * TLEN * 2);
  bf16_t* user_h = (bf16_t*)alloc(3 * MU * DIM * 2);   // later reused as attn_u (f32, smaller)
  bf16_t* news_e = (bf16_t*)alloc(3 * MN * DIM * 2);   // later reused as attn_n (f32, smaller)
  bf16_t* qu = (bf16_t*)alloc(MU * DIM * 2);
  bf16_t* ku = (bf16_t*)alloc(MU * DIM * 2);
  bf16_t* vu = (bf16_t*)alloc(MU * DIM * 2);
  bf16_t* qn = (bf16_t*)alloc(MN * DIM * 2);
  bf16_t* kn = (bf16_t*)alloc(MN * DIM * 2);
  bf16_t* vn = (bf16_t*)alloc(MN * DIM * 2);
  float*  urep = (float*)alloc((size_t)BATCH * DIM * 4);
  float*  nrep = (float*)alloc((size_t)BATCH * DIM * 4);
  bf16_t* urep_bf = (bf16_t*)alloc((size_t)BATCH * DIM * 2);
  bf16_t* nrep_bf = (bf16_t*)alloc((size_t)BATCH * DIM * 2);
  float*  logits  = (float*)alloc((size_t)BATCH * BATCH * 4);

  // 1. weight transposes -> bf16 [N,K]
  {
    const int n = DIM * DIM, g = (n + 255) / 256;
    transpose_to_bf16<<<g, 256, 0, stream>>>(Wq_n, wqn_t, DIM, DIM);
    transpose_to_bf16<<<g, 256, 0, stream>>>(Wk_n, wkn_t, DIM, DIM);
    transpose_to_bf16<<<g, 256, 0, stream>>>(Wv_n, wvn_t, DIM, DIM);
    transpose_to_bf16<<<g, 256, 0, stream>>>(Wq_u, wqu_t, DIM, DIM);
    transpose_to_bf16<<<g, 256, 0, stream>>>(Wk_u, wku_t, DIM, DIM);
    transpose_to_bf16<<<g, 256, 0, stream>>>(Wv_u, wvu_t, DIM, DIM);
    const int n2 = TLEN * DIM;
    transpose_to_bf16<<<(n2 + 255) / 256, 256, 0, stream>>>(W_u, wu_t, TLEN, DIM);
  }
  // 2. clicked titles f32 -> bf16
  {
    const size_t n = 3 * MU * TLEN;
    f32_to_bf16<<<(unsigned)((n + 255) / 256), 256, 0, stream>>>(clicked, clicked_bf, n);
  }
  // 3. embedding gather -> news_e bf16 [3*B*T, D]
  gather_embed<<<3 * BATCH * TLEN, 64, 0, stream>>>(cand, emb, news_e, 3 * BATCH * TLEN);

  // 4. user_h = clicked @ W_u + b_u   (M=153600, K=64, N=256)
  gemm_bf16<false><<<dim3((unsigned)(3 * MU / 128), 1), 256, 0, stream>>>(
      clicked_bf, wu_t, b_u, user_h, (int)(3 * MU), DIM, TLEN);

  // 5. QKV projections (each slice of dim 0 gets its own projection)
  gemm_bf16<false><<<dim3((unsigned)(MU / 128), 1), 256, 0, stream>>>(
      user_h + 0 * MU * DIM, wqu_t, bq_u, qu, (int)MU, DIM, DIM);
  gemm_bf16<false><<<dim3((unsigned)(MU / 128), 1), 256, 0, stream>>>(
      user_h + 1 * MU * DIM, wku_t, bk_u, ku, (int)MU, DIM, DIM);
  gemm_bf16<false><<<dim3((unsigned)(MU / 128), 1), 256, 0, stream>>>(
      user_h + 2 * MU * DIM, wvu_t, bv_u, vu, (int)MU, DIM, DIM);
  gemm_bf16<false><<<dim3((unsigned)(MN / 128), 1), 256, 0, stream>>>(
      news_e + 0 * MN * DIM, wqn_t, bq_n, qn, (int)MN, DIM, DIM);
  gemm_bf16<false><<<dim3((unsigned)(MN / 128), 1), 256, 0, stream>>>(
      news_e + 1 * MN * DIM, wkn_t, bk_n, kn, (int)MN, DIM, DIM);
  gemm_bf16<false><<<dim3((unsigned)(MN / 128), 1), 256, 0, stream>>>(
      news_e + 2 * MN * DIM, wvn_t, bv_n, vn, (int)MN, DIM, DIM);

  // 6. fused self-attention (scale = 1/sqrt(256) = 1/16)
  float* attn_u = (float*)user_h;   // alias: user_h dead after projections
  float* attn_n = (float*)news_e;   // alias: news_e dead after projections
  attention_kernel<HISTN><<<dim3(4, BATCH), 32, 0, stream>>>(qu, ku, vu, attn_u, 0.0625f);
  attention_kernel<TLEN ><<<dim3(4, BATCH), 32, 0, stream>>>(qn, kn, vn, attn_n, 0.0625f);

  // 7. attention pooling -> reps (f32 + bf16 copies)
  attn_pool<HISTN><<<BATCH, 256, 0, stream>>>(attn_u, aw_u, ab_u, urep, urep_bf);
  attn_pool<TLEN ><<<BATCH, 256, 0, stream>>>(attn_n, aw_n, ab_n, nrep, nrep_bf);

  // 8. logits[u,n] = user_rep[u] . news_rep[n]  (M=N=1024, K=256), then row softmax
  gemm_bf16<true><<<dim3(BATCH / 128, BATCH / 256), 256, 0, stream>>>(
      urep_bf, nrep_bf, nullptr, logits, BATCH, BATCH, DIM);
  row_softmax<<<BATCH, 256, 0, stream>>>(logits, (float*)d_out, BATCH);
}
